// CrossCorrealtion_44315472560716
// MI455X (gfx1250) — matrix operand order repaired and checked by reading, not compiled, not run on hardware
//
#include <hip/hip_runtime.h>
#include <hip/hip_bf16.h>

typedef __attribute__((ext_vector_type(16))) _Float16 v16h;
typedef _Float16 v8h   __attribute__((ext_vector_type(8), aligned(16)));
typedef _Float16 v8h_u __attribute__((ext_vector_type(8), aligned(8)));
typedef __attribute__((ext_vector_type(8)))  float    v8f;

#define DEVFN static __device__ __forceinline__

// --- CDNA5 WMMA f16 16x16x32 fragment helpers (wave32 layouts, ISA 7.12.2) ---
DEVFN int a_row(int lane)        { return lane & 15; }
DEVFN int b_col(int lane)        { return lane & 15; }
DEVFN int c_row(int lane, int r) { return r + ((lane & 16) ? 8 : 0); }
DEVFN int c_col(int lane)        { return lane & 15; }

// A fragment: row-major A, lane reads K runs [kb,kb+8) and [kb+16,kb+24)
DEVFN v16h load_a(const _Float16* __restrict__ row, int k0, int lane) {
  int kb = k0 + ((lane & 16) ? 8 : 0);
  v8h lo = *(const v8h*)(row + kb);
  v8h hi = *(const v8h*)(row + kb + 16);
  v16h a;
#pragma unroll
  for (int j = 0; j < 8; ++j) { a[j] = lo[j]; a[j + 8] = hi[j]; }
  return a;
}
// B fragment: B stored [N][K]; lane reads K runs [kb,kb+8) and [kb+16,kb+24)
DEVFN v16h load_b(const _Float16* __restrict__ row, int k0, int lane) {
  int kb = k0 + ((lane & 16) ? 8 : 0);
  v8h lo = *(const v8h*)(row + kb);
  v8h hi = *(const v8h*)(row + kb + 16);
  v16h b;
#pragma unroll
  for (int j = 0; j < 8; ++j) { b[j] = lo[j]; b[j + 8] = hi[j]; }
  return b;
}
// 8-byte aligned variant (v^T rows offset by p*100 halfwords)
DEVFN v16h load_b_u(const _Float16* __restrict__ row, int k0, int lane) {
  int kb = k0 + ((lane & 16) ? 8 : 0);
  v8h_u lo = *(const v8h_u*)(row + kb);
  v8h_u hi = *(const v8h_u*)(row + kb + 16);
  v16h b;
#pragma unroll
  for (int j = 0; j < 8; ++j) { b[j] = lo[j]; b[j + 8] = hi[j]; }
  return b;
}

DEVFN v8f wmma32(const v16h& a, const v16h& b, const v8f& c) {
  return __builtin_amdgcn_wmma_f32_16x16x32_f16(false, a, false, b, (short)0, c, false, false);
}

// Static problem config (matches reference)
#define Bq 8
#define Pn 8
#define Nn 100
#define Ln 800           // P*N
#define ROWS 6400        // B*L
#define DM 256
#define Hh 8
#define DK 32
#define Mf 64
#define Kg 16
#define NN2 10000
#define NN2P 10016       // padded K (multiple of 32)
#define PATCH 8
#define HID 128
#define FNROWS 112       // 100 padded to 7*16

// ---------------- converts ----------------
__global__ void k_cvt(const float* __restrict__ s, _Float16* __restrict__ d, int n) {
  int i = blockIdx.x * blockDim.x + threadIdx.x;
  if (i < n) d[i] = (_Float16)s[i];
}
__global__ void k_padcvt(const float* __restrict__ s, _Float16* __restrict__ d,
                         int rows, int cols, int pcols) {
  int i = blockIdx.x * blockDim.x + threadIdx.x;
  if (i >= rows * pcols) return;
  int r = i / pcols, c = i % pcols;
  d[i] = (_Float16)((c < cols) ? s[(long)r * cols + c] : 0.f);
}
// exp(gumbels) transposed: [B,L,H,K] -> egt[b][h][kk][l]
__global__ void k_egt(const float* __restrict__ g, _Float16* __restrict__ egt, int n) {
  int i = blockIdx.x * blockDim.x + threadIdx.x;
  if (i >= n) return;
  int kk = i & 15, h = (i >> 4) & 7, l = (i >> 7) % Ln, b = i / (Ln * 128);
  egt[((b * Hh + h) * Kg + kk) * Ln + l] = (_Float16)__expf(g[i]);
}

// ---------------- Kernel 1: fused QKV projection ----------------
// writes qksc f16 [6400][512] = sc*(q|k), and v transposed vt f16 [b][h][dl][l]
__global__ void k_qkv(const _Float16* __restrict__ xh, const _Float16* __restrict__ Wh,
                      const float* __restrict__ bqv, const float* __restrict__ bkv,
                      const float* __restrict__ bvv,
                      _Float16* __restrict__ qksc, _Float16* __restrict__ vt) {
  const int lane = threadIdx.x;
  const int rt = blockIdx.x;            // 0..399
  const int ct = blockIdx.y;            // 0..47
  const int tensor = ct >> 4;
  const int cw = (ct & 15) * 16;
  const float* bs = (tensor == 0) ? bqv : (tensor == 1) ? bkv : bvv;
  const int row0 = rt * 16;
  const _Float16* arow = xh + (row0 + a_row(lane)) * DM;
  const _Float16* brow = Wh + tensor * 65536 + (cw + b_col(lane)) * DM;

  v8f acc = {};
#pragma unroll
  for (int k0 = 0; k0 < DM; k0 += 32)
    acc = wmma32(load_a(arow, k0, lane), load_b(brow, k0, lane), acc);

  const float bval = bs[cw + c_col(lane)];
  if (tensor < 2) {
    const float sc = 0.84089641525f;    // (1/sqrt(tau)) * DK^-0.25
#pragma unroll
    for (int r = 0; r < 8; ++r)
      qksc[(row0 + c_row(lane, r)) * 512 + tensor * 256 + cw + c_col(lane)] =
          (_Float16)((acc[r] + bval) * sc);
  } else {
    const int d = cw + c_col(lane), h = d >> 5, dl = d & 31;
#pragma unroll
    for (int r = 0; r < 8; ++r) {
      int row = row0 + c_row(lane, r);
      int b = row / Ln, l = row % Ln;
      vt[((b * Hh + h) * DK + dl) * Ln + l] = (_Float16)(acc[r] + bval);
    }
  }
}

// ---------------- Kernel 2: performer feature map phi ----------------
__global__ void k_phi(const _Float16* __restrict__ qksc, const _Float16* __restrict__ projh,
                      _Float16* __restrict__ qph, _Float16* __restrict__ kpt) {
  __shared__ float dd[16][Mf];
  __shared__ float diag_s[16], stab_s[16];
  const int lane = threadIdx.x;
  const int rt  = blockIdx.x;
  const int h   = blockIdx.y;
  const int isq = (blockIdx.z == 0);
  const int row0 = rt * 16;
  const int bc = (isq ? 0 : 256) + h * DK;

  v16h a = load_a(qksc + (row0 + a_row(lane)) * 512 + bc, 0, lane);
#pragma unroll
  for (int nt = 0; nt < 4; ++nt) {
    v16h bb = load_b(projh + (nt * 16 + b_col(lane)) * DK, 0, lane);
    v8f c = {};
    c = wmma32(a, bb, c);
#pragma unroll
    for (int r = 0; r < 8; ++r)
      dd[c_row(lane, r)][nt * 16 + c_col(lane)] = c[r];
  }
  __syncthreads();
  if (lane < 16) {
    const _Float16* qrow = qksc + (row0 + lane) * 512 + bc;
    float dg = 0.f;
    for (int d = 0; d < DK; ++d) { float v = (float)qrow[d]; dg += v * v; }
    float st = -1e30f;
    for (int m = 0; m < Mf; ++m) st = fmaxf(st, dd[lane][m]);
    diag_s[lane] = 0.5f * dg;
    stab_s[lane] = st;   // key global-token stabilizer approximated per-row (cancels num/den)
  }
  __syncthreads();
  const float mscale = 0.125f;          // 64^-0.5
  const int b = row0 / Ln, lbase = row0 % Ln;
  for (int t = 0; t < 32; ++t) {
    int e = lane * 32 + t;
    int rr = e >> 6, m = e & 63;
    float val = mscale * (__expf(dd[rr][m] - diag_s[rr] - stab_s[rr]) + 1e-6f);
    if (isq) qph[((row0 + rr) * Hh + h) * Mf + m] = (_Float16)val;
    else     kpt[((b * Hh + h) * Mf + m) * Ln + (lbase + rr)] = (_Float16)val;
  }
}

// ---------------- Kernel 3: kvs^T = (kp*eg)^T V  + kg row sums ----------------
__global__ void k_kvs(const _Float16* __restrict__ kpt, const _Float16* __restrict__ egt,
                      const _Float16* __restrict__ vt, _Float16* __restrict__ kvst,
                      float* __restrict__ kgsum) {
  __shared__ float rowpart[32];
  const int lane = threadIdx.x;
  const int bhk = blockIdx.x;           // b*128 + h*16 + kk
  const int b = bhk >> 7, h = (bhk >> 4) & 7, kk = bhk & 15;
  const int mt = blockIdx.y;            // 0..3
  const int nt = blockIdx.z;            // 0..1
  const int m0 = mt * 16;
  const _Float16* krow = kpt + ((b * Hh + h) * Mf + m0 + a_row(lane)) * Ln;
  const _Float16* erow = egt + ((b * Hh + h) * Kg + kk) * Ln;
  const _Float16* vrow = vt  + ((b * Hh + h) * DK + nt * 16 + b_col(lane)) * Ln;

  v8f acc = {};
  float rsum = 0.f;
  for (int l0 = 0; l0 < Ln; l0 += 32) {
    int kb = l0 + ((lane & 16) ? 8 : 0);
    v8h a1 = *(const v8h*)(krow + kb) * *(const v8h*)(erow + kb);
    v8h a2 = *(const v8h*)(krow + kb + 16) * *(const v8h*)(erow + kb + 16);
    v16h a;
#pragma unroll
    for (int j = 0; j < 8; ++j) {
      a[j] = a1[j]; a[j + 8] = a2[j];
      rsum += (float)a1[j] + (float)a2[j];
    }
    acc = wmma32(a, load_b(vrow, l0, lane), acc);
  }
  const long base = ((long)(b * Hh + h) * Kg + kk);
#pragma unroll
  for (int r = 0; r < 8; ++r) {
    int m = m0 + c_row(lane, r);
    int d = nt * 16 + c_col(lane);
    kvst[(base * DK + d) * Mf + m] = (_Float16)acc[r];  // transposed: [d][m]
  }
  if (nt == 0) {                        // lanes p and p+16 share row m
    rowpart[lane] = rsum;
    __syncthreads();
    if (lane < 16)
      kgsum[base * Mf + m0 + lane] = rowpart[lane] + rowpart[lane + 16];
  }
}

// ---------------- Kernel 4: z = LN(mean_k(z_num / z_den)) ----------------
__global__ void k_z(const _Float16* __restrict__ qph, const _Float16* __restrict__ kvst,
                    const float* __restrict__ kgsum,
                    const float* __restrict__ ln_g, const float* __restrict__ ln_b,
                    float* __restrict__ z) {
  __shared__ float qs[16][Mf];
  __shared__ float den_s[16];
  __shared__ float zt[16][DK];
  __shared__ float mu_s[16], rs_s[16];
  const int lane = threadIdx.x;
  const int rt = blockIdx.x;            // tiles never straddle batches (800%16==0)
  const int h  = blockIdx.y;
  const int row0 = rt * 16;
  const int b = row0 / Ln;

  for (int t = 0; t < 32; ++t) {
    int e = lane * 32 + t;
    int rr = e >> 6, m = e & 63;
    qs[rr][m] = (float)qph[((row0 + rr) * Hh + h) * Mf + m];
  }
  __syncthreads();

  const _Float16* arow = qph + ((row0 + a_row(lane)) * Hh + h) * Mf;
  float zacc[2][8];
#pragma unroll
  for (int nt = 0; nt < 2; ++nt)
#pragma unroll
    for (int r = 0; r < 8; ++r) zacc[nt][r] = 0.f;

  for (int kk = 0; kk < Kg; ++kk) {
    const long base = ((long)(b * Hh + h) * Kg + kk);
    if (lane < 16) {
      const float* ks = &kgsum[base * Mf];
      float s = 0.f;
      for (int m = 0; m < Mf; ++m) s += qs[lane][m] * ks[m];
      den_s[lane] = s + 1e-8f;
    }
    __syncthreads();
    const _Float16* kv = kvst + base * DK * Mf;
#pragma unroll
    for (int nt = 0; nt < 2; ++nt) {
      v8f c = {};
#pragma unroll
      for (int k0 = 0; k0 < Mf; k0 += 32)
        c = wmma32(load_a(arow, k0, lane),
                   load_b(kv + (nt * 16 + b_col(lane)) * Mf, k0, lane), c);
#pragma unroll
      for (int r = 0; r < 8; ++r)
        zacc[nt][r] += c[r] / den_s[c_row(lane, r)];
    }
    __syncthreads();
  }
#pragma unroll
  for (int nt = 0; nt < 2; ++nt)
#pragma unroll
    for (int r = 0; r < 8; ++r)
      zt[c_row(lane, r)][nt * 16 + c_col(lane)] = zacc[nt][r] * (1.f / 16.f);
  __syncthreads();
  if (lane < 16) {
    float mu = 0.f;
    for (int d = 0; d < DK; ++d) mu += zt[lane][d];
    mu *= (1.f / DK);
    float s2 = 0.f;
    for (int d = 0; d < DK; ++d) { float t = zt[lane][d] - mu; s2 += t * t; }
    mu_s[lane] = mu;
    rs_s[lane] = rsqrtf(s2 * (1.f / DK) + 1e-5f);
  }
  __syncthreads();
  for (int t = 0; t < 16; ++t) {
    int e = lane * 16 + t;
    int rr = e >> 5, d = e & 31;
    z[(row0 + rr) * 256 + h * DK + d] =
        (zt[rr][d] - mu_s[rr]) * rs_s[rr] * ln_g[d] + ln_b[d];
  }
}

// ---------------- Kernel 5: adjacency FC (512 x 10016 x 128) + ReLU ----------------
__global__ void k_h1(const _Float16* __restrict__ adjh, const _Float16* __restrict__ fcwh,
                     const float* __restrict__ fcb, float* __restrict__ h1) {
  const int lane = threadIdx.x;
  const int rt = blockIdx.x, ct = blockIdx.y;
  const int row0 = rt * 16, col0 = ct * 16;
  const _Float16* arow = adjh + (long)(row0 + a_row(lane)) * NN2P;
  const _Float16* brow = fcwh + (long)(col0 + b_col(lane)) * NN2P;
  v8f acc = {};
  for (int k0 = 0; k0 < NN2P; k0 += 32) {
    __builtin_prefetch(brow + k0 + 128, 0, 1);
    acc = wmma32(load_a(arow, k0, lane), load_b(brow, k0, lane), acc);
  }
#pragma unroll
  for (int r = 0; r < 8; ++r) {
    float v = acc[r] + fcb[col0 + c_col(lane)];
    h1[(row0 + c_row(lane, r)) * HID + col0 + c_col(lane)] = fmaxf(v, 0.f);
  }
}

// ---------------- Kernel 6: logits + softmax over patch ----------------
__global__ void k_softw(const float* __restrict__ h1, const float* __restrict__ ow,
                        const float* __restrict__ ob, float* __restrict__ w) {
  __shared__ float lg[PATCH];
  const int lane = threadIdx.x;
  const int bp = blockIdx.x;
  if (lane < PATCH) {
    const float* hp = &h1[(bp * PATCH + lane) * HID];
    float s = ob[0];
    for (int i = 0; i < HID; ++i) s += hp[i] * ow[i];
    lg[lane] = s;
  }
  __syncthreads();
  if (lane == 0) {
    float mx = lg[0];
    for (int i = 1; i < PATCH; ++i) mx = fmaxf(mx, lg[i]);
    float e[PATCH], se = 0.f;
    for (int i = 0; i < PATCH; ++i) { e[i] = __expf(lg[i] - mx); se += e[i]; }
    for (int i = 0; i < PATCH; ++i) w[bp * PATCH + i] = e[i] / se;
  }
}

// ---------------- Kernel 7: fused adj + L1 normalize -> fnh f16 [64][112][128] ----------------
__global__ void k_fused(const float* __restrict__ adj, const float* __restrict__ w,
                        _Float16* __restrict__ fnh) {
  __shared__ float red[128];
  const int t = threadIdx.x;
  const int blk = blockIdx.x;           // bp*112 + i
  const int bp = blk / FNROWS, i = blk % FNROWS;
  _Float16* orow = fnh + (long)blk * HID;
  if (i >= Nn) { orow[t] = (_Float16)0.f; return; }   // uniform branch per block
  const int b = bp >> 3, p = bp & 7;
  float val = 0.f, asum = 0.f;
  if (t < Nn) {
    float s = 0.f;
    for (int pl = 0; pl < PATCH; ++pl)
      s += adj[((long)(b * 64 + p * PATCH + pl) * Nn + i) * Nn + t] * w[bp * PATCH + pl];
    val = s;
    asum = fabsf(s);
  }
  red[t] = asum;
  __syncthreads();
  for (int off = 64; off > 0; off >>= 1) {
    if (t < off) red[t] += red[t + off];
    __syncthreads();
  }
  float denom = fmaxf(red[0], 1e-12f);
  orow[t] = (_Float16)((t < Nn) ? val / denom : 0.f);
}

// ---------------- Kernel 8: relational bias + add z -> zbh f16 ----------------
__global__ void k_bias(const _Float16* __restrict__ fnh, const _Float16* __restrict__ vt,
                       const float* __restrict__ z, _Float16* __restrict__ zbh) {
  const int lane = threadIdx.x;
  const int bp = blockIdx.x;            // 0..63
  const int mt = blockIdx.y;            // 0..6
  const int nt = blockIdx.z;            // 0..15
  const int b = bp >> 3, p = bp & 7;
  const int m0 = mt * 16, n0 = nt * 16;
  const _Float16* arow = fnh + (long)(bp * FNROWS + m0 + a_row(lane)) * HID;
  const int d = n0 + b_col(lane), h = d >> 5, dl = d & 31;
  const _Float16* brow = vt + ((b * Hh + h) * DK + dl) * Ln + p * Nn;

  v8f acc = {};
#pragma unroll
  for (int k0 = 0; k0 < HID; k0 += 32)   // padded A rows are zero beyond k=99
    acc = wmma32(load_a(arow, k0, lane), load_b_u(brow, k0, lane), acc);

#pragma unroll
  for (int r = 0; r < 8; ++r) {
    int m = m0 + c_row(lane, r);
    if (m < Nn) {
      int gr = b * Ln + p * Nn + m;
      zbh[gr * 256 + n0 + c_col(lane)] =
          (_Float16)(z[gr * 256 + n0 + c_col(lane)] + acc[r]);
    }
  }
}

// ---------------- Kernel 9: output projection -> d_out f32 ----------------
__global__ void k_out(const _Float16* __restrict__ zbh, const _Float16* __restrict__ Woh,
                      const float* __restrict__ bo, float* __restrict__ out) {
  const int lane = threadIdx.x;
  const int rt = blockIdx.x, ct = blockIdx.y;
  const int row0 = rt * 16, col0 = ct * 16;
  const _Float16* arow = zbh + (row0 + a_row(lane)) * DM;
  const _Float16* brow = Woh + (col0 + b_col(lane)) * DM;
  v8f acc = {};
#pragma unroll
  for (int k0 = 0; k0 < DM; k0 += 32)
    acc = wmma32(load_a(arow, k0, lane), load_b(brow, k0, lane), acc);
#pragma unroll
  for (int r = 0; r < 8; ++r)
    out[(row0 + c_row(lane, r)) * DM + col0 + c_col(lane)] = acc[r] + bo[col0 + c_col(lane)];
}

extern "C" void kernel_launch(void* const* d_in, const int* in_sizes, int n_in,
                              void* d_out, int out_size, void* d_ws, size_t ws_size,
                              hipStream_t stream) {
  const float* x    = (const float*)d_in[0];
  const float* adj  = (const float*)d_in[1];
  const float* Wq   = (const float*)d_in[2];
  const float* bqv  = (const float*)d_in[3];
  const float* Wk   = (const float*)d_in[4];
  const float* bkv  = (const float*)d_in[5];
  const float* Wv   = (const float*)d_in[6];
  const float* bvv  = (const float*)d_in[7];
  const float* Wo   = (const float*)d_in[8];
  const float* bo   = (const float*)d_in[9];
  const float* ln_g = (const float*)d_in[10];
  const float* ln_b = (const float*)d_in[11];
  const float* fcw  = (const float*)d_in[12];
  const float* fcb  = (const float*)d_in[13];
  const float* ow   = (const float*)d_in[14];
  const float* ob   = (const float*)d_in[15];
  const float* proj = (const float*)d_in[16];
  const float* gmb  = (const float*)d_in[17];
  (void)in_sizes; (void)n_in; (void)out_size; (void)ws_size;

  char* base = (char*)d_ws;
  size_t off = 0;
  auto allocB = [&](size_t bytes) {
    void* p = base + off;
    off = (off + bytes + 255) & ~(size_t)255;
    return p;
  };
  _Float16* xh    = (_Float16*)allocB((size_t)ROWS * DM * 2);
  _Float16* Wh    = (_Float16*)allocB((size_t)3 * 65536 * 2);
  _Float16* Woh   = (_Float16*)allocB((size_t)65536 * 2);
  _Float16* projh = (_Float16*)allocB((size_t)Mf * DK * 2);
  _Float16* adjh  = (_Float16*)allocB((size_t)512 * NN2P * 2);
  _Float16* fcwh  = (_Float16*)allocB((size_t)HID * NN2P * 2);
  _Float16* qksc  = (_Float16*)allocB((size_t)ROWS * 512 * 2);
  _Float16* vt    = (_Float16*)allocB(((size_t)Bq * Hh * DK * Ln + 128) * 2); // +slack: padded-K reads
  _Float16* egt   = (_Float16*)allocB((size_t)Bq * Hh * Kg * Ln * 2);
  _Float16* qph   = (_Float16*)allocB((size_t)ROWS * Hh * Mf * 2);
  _Float16* kpt   = (_Float16*)allocB((size_t)Bq * Hh * Mf * Ln * 2);
  _Float16* kvst  = (_Float16*)allocB((size_t)Bq * Hh * Kg * DK * Mf * 2);
  float*    kgsum = (float*)allocB((size_t)Bq * Hh * Kg * Mf * 4);
  float*    zbuf  = (float*)allocB((size_t)ROWS * DM * 4);
  float*    h1    = (float*)allocB((size_t)512 * HID * 4);
  float*    wsm   = (float*)allocB((size_t)64 * PATCH * 4);
  _Float16* fnh   = (_Float16*)allocB((size_t)64 * FNROWS * HID * 2);
  _Float16* zbh   = (_Float16*)allocB((size_t)ROWS * DM * 2);

  // f32 -> f16 staging (once per tensor)
  k_cvt<<<(ROWS * DM + 255) / 256, 256, 0, stream>>>(x, xh, ROWS * DM);
  k_cvt<<<(65536 + 255) / 256, 256, 0, stream>>>(Wq, Wh, 65536);
  k_cvt<<<(65536 + 255) / 256, 256, 0, stream>>>(Wk, Wh + 65536, 65536);
  k_cvt<<<(65536 + 255) / 256, 256, 0, stream>>>(Wv, Wh + 131072, 65536);
  k_cvt<<<(65536 + 255) / 256, 256, 0, stream>>>(Wo, Woh, 65536);
  k_cvt<<<(Mf * DK + 255) / 256, 256, 0, stream>>>(proj, projh, Mf * DK);
  k_padcvt<<<(512 * NN2P + 255) / 256, 256, 0, stream>>>(adj, adjh, 512, NN2, NN2P);
  k_padcvt<<<(HID * NN2P + 255) / 256, 256, 0, stream>>>(fcw, fcwh, HID, NN2, NN2P);
  k_egt<<<(ROWS * Hh * Kg + 255) / 256, 256, 0, stream>>>(gmb, egt, ROWS * Hh * Kg);

  k_qkv  <<<dim3(400, 48),    32, 0, stream>>>(xh, Wh, bqv, bkv, bvv, qksc, vt);
  k_phi  <<<dim3(400, 8, 2),  32, 0, stream>>>(qksc, projh, qph, kpt);
  k_kvs  <<<dim3(1024, 4, 2), 32, 0, stream>>>(kpt, egt, vt, kvst, kgsum);
  k_z    <<<dim3(400, 8),     32, 0, stream>>>(qph, kvst, kgsum, ln_g, ln_b, zbuf);
  k_h1   <<<dim3(32, 8),      32, 0, stream>>>(adjh, fcwh, fcb, h1);
  k_softw<<<64,               32, 0, stream>>>(h1, ow, ob, wsm);
  k_fused<<<64 * FNROWS,     128, 0, stream>>>(adj, wsm, fnh);
  k_bias <<<dim3(64, 7, 16),  32, 0, stream>>>(fnh, vt, zbuf, zbh);
  k_out  <<<dim3(400, 16),    32, 0, stream>>>(zbh, Woh, bo, (float*)d_out);
}
